// RNO_layer_15788299780726
// MI455X (gfx1250) — compile-verified, matrix operand order checked by reading, not run on hardware
//
#include <hip/hip_runtime.h>
#include <hip/hip_bf16.h>
#include <cstdint>
#include <cstddef>

// ---------------------------------------------------------------------------
// FNO-GRU for MI455X (gfx1250): bf16 WMMA (f32 accumulate) for all channel
// mixing, fragment-ready packed weights, 16-row padded accumulators
// (branch-free WMMA tiles), LDS-staged spectral transforms, wave32 layouts
// per CDNA5 ISA 7.12.2.
// ---------------------------------------------------------------------------

#define B_    4
#define T_    8
#define C_    64
#define HW_   4096
#define BCHW_ (B_ * C_ * HW_)     // 1,048,576
#define NMODE_ 512                 // 2 regions x 16 x 16
#define TWO_PI 6.28318530717958647692f

// padded per-mode matrices: 16 rows (WMMA M), 128 cols (re|im channels)
#define AMODE_ELEMS ((size_t)NMODE_ * 16 * 128)            // ushort
#define YMODE_ELEMS ((size_t)NMODE_ * 16 * 128)            // float

// packed spectral weights: [6][512 modes][8 nt][4 kc][32 lane][16 e] ushort
#define WMODE_PER_BLK ((size_t)NMODE_ * 8 * 4 * 32 * 16)   // 8,388,608
#define WMODE_TOT     (6 * WMODE_PER_BLK)                  // 50,331,648
// packed skip weights: [6][4 nt][2 kc][32 lane][16 e] ushort
#define WSKIP_PER_BLK ((size_t)4 * 2 * 32 * 16)            // 4096
#define WSKIP_TOT     (6 * WSKIP_PER_BLK)                  // 24,576

typedef __attribute__((ext_vector_type(16))) __bf16 v16bf;
typedef __attribute__((ext_vector_type(8)))  float  v8f;

union V16 { v16bf v; unsigned short u[16]; };

__device__ __forceinline__ unsigned short f2bf(float f) {
  unsigned int u = __float_as_uint(f);
  u += 0x7FFFu + ((u >> 16) & 1u);       // round-to-nearest-even
  return (unsigned short)(u >> 16);
}

// CDNA5 16-bit A/B fragment K index for vector element e, lane-half h.
// (ISA 7.12.2: VGPR v<4 -> K=2v,2v+1 ; v>=4 -> K=16+2(v-4).. ; +8 for lanes 16-31)
__device__ __forceinline__ int wmma_k16(int e, int half) {
  int v = e >> 1, odd = e & 1;
  int base = (v < 4) ? (2 * v) : (16 + 2 * (v - 4));
  return base + 8 * half + odd;
}

// --------------------------- tiny utility kernels --------------------------
__global__ void k_init_h(const float* __restrict__ bias_h, float* __restrict__ h) {
  size_t i = (size_t)blockIdx.x * blockDim.x + threadIdx.x;
  if (i < (size_t)BCHW_) h[i] = bias_h[0];
}

__global__ void k_zero_u16(unsigned short* __restrict__ p, size_t n) {
  size_t i = (size_t)blockIdx.x * blockDim.x + threadIdx.x;
  if (i < n) p[i] = 0;
}

// ---- pack spectral weights into WMMA B-fragment layout (once / launch) ----
// Real-ified complex GEMM: B[gk][gn] with gk = re(i)|im(i+64), gn = re(o)|im(o+64)
//   B = [[wr, wi], [-wi, wr]]  (sign baked in here)
__global__ void k_pack_wmode(const float* __restrict__ sw1,
                             const float* __restrict__ sw2,
                             unsigned short* __restrict__ W) {
  size_t idx = (size_t)blockIdx.x * blockDim.x + threadIdx.x;
  if (idx >= WMODE_TOT) return;
  int e    = (int)(idx & 15);  size_t r = idx >> 4;
  int lane = (int)(r & 31);    r >>= 5;
  int kc   = (int)(r & 3);     r >>= 2;
  int nt   = (int)(r & 7);     r >>= 3;
  int m    = (int)(r & 511);   r >>= 9;
  int wb   = (int)r;                                   // 0..5
  int region = m >> 8, u = (m >> 4) & 15, wy = m & 15;
  int kl = wmma_k16(e, lane >> 4);
  int gk = kc * 32 + kl;
  int gn = nt * 16 + (lane & 15);
  int i = gk & 63, o = gn & 63;
  const float2* src = (const float2*)((region ? sw2 : sw1) +
                                      (size_t)wb * C_ * C_ * 16 * 16 * 2);
  float2 wv = src[(((size_t)i * 64 + o) * 16 + u) * 16 + wy];
  float val = (gk < 64) ? ((gn < 64) ? wv.x : wv.y)
                        : ((gn < 64) ? -wv.y : wv.x);
  W[idx] = f2bf(val);
}

// ---- pack skip weights into WMMA B-fragment layout ------------------------
__global__ void k_pack_skw(const float* __restrict__ skw,
                           unsigned short* __restrict__ W) {
  size_t idx = (size_t)blockIdx.x * blockDim.x + threadIdx.x;
  if (idx >= WSKIP_TOT) return;
  int e    = (int)(idx & 15);  size_t r = idx >> 4;
  int lane = (int)(r & 31);    r >>= 5;
  int kc   = (int)(r & 1);     r >>= 1;
  int nt   = (int)(r & 3);     r >>= 2;
  int blk  = (int)r;
  int kl = wmma_k16(e, lane >> 4);
  int gk = kc * 32 + kl;
  int gn = nt * 16 + (lane & 15);
  W[idx] = f2bf(skw[(size_t)blk * C_ * C_ + (size_t)gk * 64 + gn]);
}

// ----------------- [B,C,HW] f32 -> [B,HW,C] bf16 (LDS tiled) ---------------
__global__ void k_transpose(const float* __restrict__ in, size_t bstride,
                            unsigned short* __restrict__ out) {
  __shared__ float tile[64 * 65];
  int b   = blockIdx.y;
  int hw0 = blockIdx.x * 64;
  for (int i = threadIdx.x; i < 64 * 64; i += blockDim.x) {
    int c = i >> 6, hw = i & 63;                           // coalesced along hw
    tile[c * 65 + hw] = in[(size_t)b * bstride + (size_t)c * HW_ + hw0 + hw];
  }
  __syncthreads();
  for (int i = threadIdx.x; i < 64 * 64; i += blockDim.x) {
    int hw = i >> 6, c = i & 63;                           // coalesced along c
    out[((size_t)b * HW_ + hw0 + hw) * 64 + c] = f2bf(tile[c * 65 + hw]);
  }
}

// ------------- forward spectral transform -> WMMA A layout -----------------
// One block per (b,c) image. Output: Amode[m][b][k] bf16 (rows 4..15 stay 0).
__global__ void k_fwd_dft(const float* __restrict__ in, size_t bstride,
                          unsigned short* __restrict__ Amode) {
  int img = blockIdx.x;
  int b = img >> 6, c = img & 63;
  const float* src = in + (size_t)b * bstride + (size_t)c * HW_;

  __shared__ float s_img[64 * 64];
  __shared__ float s_re[64 * 16], s_im[64 * 16];
  __shared__ float twc[64], tws[64];

  if (threadIdx.x < 64) {
    float s, co;
    __sincosf(TWO_PI * (float)threadIdx.x * (1.0f / 64.0f), &s, &co);
    twc[threadIdx.x] = co; tws[threadIdx.x] = s;
  }
  for (int i = threadIdx.x; i < 4096; i += blockDim.x) s_img[i] = src[i];
  __syncthreads();

  // stage 1: per-row DFT, W bins 0..15, e^{-i a}
  for (int i = threadIdx.x; i < 64 * 16; i += blockDim.x) {
    int row = i >> 4, k = i & 15;
    float re = 0.f, im = 0.f;
    for (int n = 0; n < 64; ++n) {
      int idx = (n * k) & 63;
      float v = s_img[row * 64 + n];
      re += v * twc[idx];
      im -= v * tws[idx];
    }
    s_re[i] = re; s_im[i] = im;
  }
  __syncthreads();

  // stage 2: column DFT, keep H rows {0..15, 48..63}; ortho scale 1/64
  for (int i = threadIdx.x; i < NMODE_; i += blockDim.x) {
    int region = i >> 8, u = (i >> 4) & 15, k = i & 15;
    int fr = region ? (48 + u) : u;
    float re = 0.f, im = 0.f;
    for (int n = 0; n < 64; ++n) {
      int idx = (n * fr) & 63;
      float co = twc[idx], s = tws[idx];
      float ar = s_re[n * 16 + k], ai = s_im[n * 16 + k];
      re += ar * co + ai * s;    // (ar + i ai) * (co - i s)
      im += ai * co - ar * s;
    }
    re *= (1.f / 64.f); im *= (1.f / 64.f);
    unsigned short* dst = Amode + ((size_t)i * 16 + b) * 128;
    dst[c]      = f2bf(re);
    dst[64 + c] = f2bf(im);
  }
}

// --------- per-mode complex channel mix: WMMA bf16 16x16x32 ----------------
// One wave per mode. Real-ified GEMM: [16(M) x 128(K)] x [128(K) x 128(N)].
// A and Y are 16-row padded (pad rows are exact zeros) -> fully branch-free.
__global__ __launch_bounds__(256) void k_mode_mix(
    const unsigned short* __restrict__ Amode,  // [512][16][128]
    const unsigned short* __restrict__ Wm,     // block base, fragment layout
    float* __restrict__ Y,                     // [512][16][128]
    int accumulate) {
  int wid  = threadIdx.x >> 5;
  int lane = threadIdx.x & 31;
  int m    = blockIdx.x * 8 + wid;               // 0..511
  int half = lane >> 4, lo = lane & 15;

  const unsigned short* wbase = Wm + (size_t)m * (8 * 4 * 512) + (size_t)lane * 16;
  __builtin_prefetch((const void*)wbase, 0, 1);

  // A fragments: unconditional, lane-contiguous ushort pairs
  V16 A[4];
  const unsigned short* arow = Amode + ((size_t)m * 16 + lo) * 128;
#pragma unroll
  for (int kc = 0; kc < 4; ++kc)
#pragma unroll
    for (int e = 0; e < 16; ++e) {
      int kl = wmma_k16(e, half);
      A[kc].u[e] = arow[kc * 32 + kl];
    }

  float* yrow = Y + (size_t)m * 16 * 128;

  for (int nt = 0; nt < 8; ++nt) {
    // hoist all 4 B tiles: one clause of 8 x global_load_b128
    V16 Bt[4];
#pragma unroll
    for (int kc = 0; kc < 4; ++kc)
      Bt[kc] = *(const V16*)(wbase + ((size_t)nt * 4 + kc) * 512);

    v8f Cacc = {};
    if (accumulate) {
#pragma unroll
      for (int g = 0; g < 8; ++g)
        Cacc[g] = yrow[(size_t)(g + 8 * half) * 128 + nt * 16 + lo];
    }
#pragma unroll
    for (int kc = 0; kc < 4; ++kc)
      Cacc = __builtin_amdgcn_wmma_f32_16x16x32_bf16(
          false, A[kc].v, false, Bt[kc].v, (short)0, Cacc, false, false);

#pragma unroll
    for (int g = 0; g < 8; ++g)
      yrow[(size_t)(g + 8 * half) * 128 + nt * 16 + lo] = Cacc[g];
  }
}

// ------------------- inverse spectral transform ----------------------------
// One block per (b,o) image: Y modes -> spatial [B,C,H,W] f32.
__global__ void k_irfft(const float* __restrict__ Y, float* __restrict__ S) {
  int img = blockIdx.x;
  int b = img >> 6, o = img & 63;
  __shared__ float fr_re[32 * 16], fr_im[32 * 16];
  __shared__ float t_re[64 * 16],  t_im[64 * 16];
  __shared__ float twc[64], tws[64];

  if (threadIdx.x < 64) {
    float s, co;
    __sincosf(TWO_PI * (float)threadIdx.x * (1.0f / 64.0f), &s, &co);
    twc[threadIdx.x] = co; tws[threadIdx.x] = s;
  }
  for (int i = threadIdx.x; i < NMODE_; i += blockDim.x) {
    const float* src = Y + ((size_t)i * 16 + b) * 128;
    fr_re[i] = src[o];
    fr_im[i] = src[64 + o];
  }
  __syncthreads();

  // inverse along H: 32 stored rows -> 64 spatial rows, e^{+i a}
  for (int i = threadIdx.x; i < 64 * 16; i += blockDim.x) {
    int n = i >> 4, k = i & 15;
    float re = 0.f, im = 0.f;
    for (int r = 0; r < 32; ++r) {
      int fr = (r < 16) ? r : (r + 32);
      int idx = (fr * n) & 63;
      float co = twc[idx], s = tws[idx];
      float ar = fr_re[r * 16 + k], ai = fr_im[r * 16 + k];
      re += ar * co - ai * s;
      im += ar * s + ai * co;
    }
    t_re[i] = re; t_im[i] = im;
  }
  __syncthreads();

  // inverse real transform along W (bins 16..48 are zero -> Hermitian x2)
  for (int i = threadIdx.x; i < 4096; i += blockDim.x) {
    int n = i >> 6, w = i & 63;
    float acc = t_re[n * 16 + 0];
    for (int k = 1; k < 16; ++k) {
      int idx = (k * w) & 63;
      acc += 2.f * (t_re[n * 16 + k] * twc[idx] - t_im[n * 16 + k] * tws[idx]);
    }
    S[((size_t)b * 64 + o) * HW_ + i] = acc * (1.f / 64.f);
  }
}

// ------------------ skip 1x1 conv: 16384x64x64 WMMA GEMM -------------------
__global__ __launch_bounds__(256) void k_skip_gemm(
    const unsigned short* __restrict__ act,  // [16384][64] bf16 rows (b,hw)
    const unsigned short* __restrict__ Wk,   // block base, fragment layout
    float* __restrict__ G) {                 // [16384][64] f32
  int wid  = threadIdx.x >> 5;
  int lane = threadIdx.x & 31;
  int tile = blockIdx.x * 8 + wid;           // 0..1023
  int row0 = tile * 16;
  int half = lane >> 4, lo = lane & 15;

  const unsigned short* wbase = Wk + (size_t)lane * 16;

  V16 A[2];
  const unsigned short* arow = act + (size_t)(row0 + lo) * 64;
#pragma unroll
  for (int kc = 0; kc < 2; ++kc)
#pragma unroll
    for (int e = 0; e < 16; ++e) {
      int kl = wmma_k16(e, half);
      A[kc].u[e] = arow[kc * 32 + kl];
    }

  for (int nt = 0; nt < 4; ++nt) {
    V16 Bt[2];
#pragma unroll
    for (int kc = 0; kc < 2; ++kc)
      Bt[kc] = *(const V16*)(wbase + ((size_t)nt * 2 + kc) * 512);

    v8f Cacc = {};
#pragma unroll
    for (int kc = 0; kc < 2; ++kc)
      Cacc = __builtin_amdgcn_wmma_f32_16x16x32_bf16(
          false, A[kc].v, false, Bt[kc].v, (short)0, Cacc, false, false);

#pragma unroll
    for (int g = 0; g < 8; ++g) {
      int M = g + 8 * half;
      G[(size_t)(row0 + M) * 64 + nt * 16 + lo] = Cacc[g];
    }
  }
}

// ------------------------- GRU gate kernels --------------------------------
__global__ void k_gates(const float* __restrict__ Sz, const float* __restrict__ Sr,
                        const float* __restrict__ G0, const float* __restrict__ G1,
                        const float* __restrict__ G2, const float* __restrict__ G3,
                        const float* __restrict__ gate_b, const float* __restrict__ h,
                        float* __restrict__ zb, float* __restrict__ rh) {
  size_t i = (size_t)blockIdx.x * blockDim.x + threadIdx.x;
  if (i >= (size_t)BCHW_) return;
  int hw = (int)(i & 4095);
  size_t bo = i >> 12;
  int o = (int)(bo & 63), b = (int)(bo >> 6);
  size_t g = ((size_t)b * HW_ + hw) * 64 + o;
  float z = Sz[i] + G0[g] + G1[g] + gate_b[0];
  z = 1.f / (1.f + __expf(-z));
  float r = Sr[i] + G2[g] + G3[g] + gate_b[1];
  r = 1.f / (1.f + __expf(-r));
  zb[i] = z;
  rh[i] = r * h[i];
}

__global__ void k_update(const float* __restrict__ Sh,
                         const float* __restrict__ G4, const float* __restrict__ G5,
                         const float* __restrict__ gate_b,
                         const float* __restrict__ zb, const float* __restrict__ h,
                         float* __restrict__ dst) {
  size_t i = (size_t)blockIdx.x * blockDim.x + threadIdx.x;
  if (i >= (size_t)BCHW_) return;
  int hw = (int)(i & 4095);
  size_t bo = i >> 12;
  int o = (int)(bo & 63), b = (int)(bo >> 6);
  size_t g = ((size_t)b * HW_ + hw) * 64 + o;
  float v = Sh[i] + G4[g] + G5[g] + gate_b[2];
  const float alpha = 1.6732632423543772f, scale = 1.0507009873554805f;
  float hh = scale * (v > 0.f ? v : alpha * (__expf(v) - 1.f));
  float z = zb[i];
  dst[i] = (1.f - z) * h[i] + z * hh;
}

// ---------------------------------------------------------------------------
extern "C" void kernel_launch(void* const* d_in, const int* in_sizes, int n_in,
                              void* d_out, int out_size, void* d_ws, size_t ws_size,
                              hipStream_t stream) {
  const float* x      = (const float*)d_in[0];   // [B,T,C,H,W]
  const float* sw1    = (const float*)d_in[1];   // [6,C,C,16,16,2]
  const float* sw2    = (const float*)d_in[2];
  const float* skw    = (const float*)d_in[3];   // [6,C,C]
  const float* gate_b = (const float*)d_in[4];   // [3]
  const float* bias_h = (const float*)d_in[5];   // [1]
  float* out = (float*)d_out;

  // ---- carve workspace ----
  char* p = (char*)d_ws;
  auto alloc = [&](size_t bytes) -> void* {
    void* r = (void*)p;
    p += (bytes + 255) & ~(size_t)255;
    return r;
  };
  const size_t fBytes = (size_t)BCHW_ * sizeof(float);          // 4 MB
  float* h   = (float*)alloc(fBytes);
  float* zb  = (float*)alloc(fBytes);
  float* rh  = (float*)alloc(fBytes);
  float* Sz  = (float*)alloc(fBytes);
  float* Sr  = (float*)alloc(fBytes);
  float* Sh  = (float*)alloc(fBytes);
  float* G0  = (float*)alloc(fBytes);
  float* G1  = (float*)alloc(fBytes);
  float* G2  = (float*)alloc(fBytes);
  float* G3  = (float*)alloc(fBytes);
  float* G4  = (float*)alloc(fBytes);
  float* G5  = (float*)alloc(fBytes);
  unsigned short* xtT = (unsigned short*)alloc((size_t)B_ * HW_ * 64 * 2);
  unsigned short* hT  = (unsigned short*)alloc((size_t)B_ * HW_ * 64 * 2);
  unsigned short* rhT = (unsigned short*)alloc((size_t)B_ * HW_ * 64 * 2);
  unsigned short* Ax  = (unsigned short*)alloc(AMODE_ELEMS * 2);
  unsigned short* Ah  = (unsigned short*)alloc(AMODE_ELEMS * 2);
  unsigned short* Arh = (unsigned short*)alloc(AMODE_ELEMS * 2);
  float* Yz = (float*)alloc(YMODE_ELEMS * sizeof(float));
  float* Yr = (float*)alloc(YMODE_ELEMS * sizeof(float));
  float* Yh = (float*)alloc(YMODE_ELEMS * sizeof(float));
  unsigned short* Wm = (unsigned short*)alloc(WMODE_TOT * 2);   // ~100 MB
  unsigned short* Wk = (unsigned short*)alloc(WSKIP_TOT * 2);
  if ((size_t)(p - (char*)d_ws) > ws_size) return;   // not enough scratch

  const size_t hbs = (size_t)C_ * HW_;       // batch stride of [B,C,H,W]
  const size_t xbs = (size_t)T_ * C_ * HW_;  // batch stride inside x
  dim3 tg(64, B_);
  const int eg = BCHW_ / 256;

  // ---- once per launch: pack weights, zero padded A buffers, init h ----
  k_pack_wmode<<<(unsigned)((WMODE_TOT + 255) / 256), 256, 0, stream>>>(sw1, sw2, Wm);
  k_pack_skw<<<(unsigned)((WSKIP_TOT + 255) / 256), 256, 0, stream>>>(skw, Wk);
  {
    unsigned zg = (unsigned)((AMODE_ELEMS + 255) / 256);
    k_zero_u16<<<zg, 256, 0, stream>>>(Ax, AMODE_ELEMS);
    k_zero_u16<<<zg, 256, 0, stream>>>(Ah, AMODE_ELEMS);
    k_zero_u16<<<zg, 256, 0, stream>>>(Arh, AMODE_ELEMS);
  }
  k_init_h<<<eg, 256, 0, stream>>>(bias_h, h);

  auto wmb = [&](int i) { return Wm + (size_t)i * WMODE_PER_BLK; };
  auto wkb = [&](int i) { return Wk + (size_t)i * WSKIP_PER_BLK; };

  for (int t = 0; t < T_; ++t) {
    const float* xt = x + (size_t)t * C_ * HW_;

    // ---- phase A: z, r (and xt contribution to h_hat) ----
    k_transpose<<<tg, 256, 0, stream>>>(xt, xbs, xtT);
    k_transpose<<<tg, 256, 0, stream>>>(h, hbs, hT);
    k_fwd_dft<<<B_ * C_, 128, 0, stream>>>(xt, xbs, Ax);
    k_fwd_dft<<<B_ * C_, 128, 0, stream>>>(h, hbs, Ah);

    k_mode_mix<<<64, 256, 0, stream>>>(Ax, wmb(0), Yz, 0);
    k_mode_mix<<<64, 256, 0, stream>>>(Ah, wmb(1), Yz, 1);
    k_mode_mix<<<64, 256, 0, stream>>>(Ax, wmb(2), Yr, 0);
    k_mode_mix<<<64, 256, 0, stream>>>(Ah, wmb(3), Yr, 1);
    k_mode_mix<<<64, 256, 0, stream>>>(Ax, wmb(4), Yh, 0);

    k_irfft<<<B_ * C_, 128, 0, stream>>>(Yz, Sz);
    k_irfft<<<B_ * C_, 128, 0, stream>>>(Yr, Sr);

    k_skip_gemm<<<128, 256, 0, stream>>>(xtT, wkb(0), G0);
    k_skip_gemm<<<128, 256, 0, stream>>>(hT,  wkb(1), G1);
    k_skip_gemm<<<128, 256, 0, stream>>>(xtT, wkb(2), G2);
    k_skip_gemm<<<128, 256, 0, stream>>>(hT,  wkb(3), G3);
    k_skip_gemm<<<128, 256, 0, stream>>>(xtT, wkb(4), G4);

    k_gates<<<eg, 256, 0, stream>>>(Sz, Sr, G0, G1, G2, G3, gate_b, h, zb, rh);

    // ---- phase B: h_hat via f(5, r*h), then GRU update ----
    k_transpose<<<tg, 256, 0, stream>>>(rh, hbs, rhT);
    k_fwd_dft<<<B_ * C_, 128, 0, stream>>>(rh, hbs, Arh);
    k_mode_mix<<<64, 256, 0, stream>>>(Arh, wmb(5), Yh, 1);
    k_irfft<<<B_ * C_, 128, 0, stream>>>(Yh, Sh);
    k_skip_gemm<<<128, 256, 0, stream>>>(rhT, wkb(5), G5);

    float* dst = (t == T_ - 1) ? out : h;
    k_update<<<eg, 256, 0, stream>>>(Sh, G4, G5, gate_b, zb, h, dst);
  }
}
